// RNN_55482387530426
// MI455X (gfx1250) — matrix operand order repaired and checked by reading, not compiled, not run on hardware
//
#include <hip/hip_runtime.h>
#include <hip/hip_bf16.h>
#include <stdint.h>

typedef __bf16 bf16_t;
typedef __bf16 v8bf  __attribute__((ext_vector_type(8)));
typedef __bf16 v16bf __attribute__((ext_vector_type(16)));
typedef float  v8f   __attribute__((ext_vector_type(8)));

#define HID   256
#define SEQT  32
#define NCLS  12
#define NBATCH 16384
#define NROWS (NBATCH * SEQT)   // 524288

// ---------- helpers ----------
__device__ __forceinline__ bf16_t f32_to_bf16(float f) {
    union { float f; uint32_t u; } v; v.f = f;
    uint32_t u = v.u;
    uint32_t r = u + 0x7FFFu + ((u >> 16) & 1u);   // round-to-nearest-even
    union { uint16_t s; bf16_t b; } o; o.s = (uint16_t)(r >> 16);
    return o.b;
}

// ---------- elementwise conversion ----------
__global__ __launch_bounds__(256) void cvt_f32_bf16(const float* __restrict__ src,
                                                    bf16_t* __restrict__ dst, int n) {
    int i = blockIdx.x * 256 + threadIdx.x;
    if (i < n) dst[i] = f32_to_bf16(src[i]);
}

__global__ __launch_bounds__(256) void add_bias_vec(const float* __restrict__ a,
                                                    const float* __restrict__ b,
                                                    float* __restrict__ out, int n) {
    int i = blockIdx.x * 256 + threadIdx.x;
    if (i < n) out[i] = a[i] + b[i];
}

// ---------- xw = A[R,K] * W[256,K]^T + bias, f32 out ----------
// Block: 256 threads (8 waves). Tile: 128 rows x 256 cols.
// K is a COMPILE-TIME constant so the per-K fragment array stays in VGPRs
// (runtime K caused scratch spills of afrag[] in the previous round).
template <int K>
__global__ __launch_bounds__(256) void wmma_gemm_xw(const bf16_t* __restrict__ A,
                                                    const bf16_t* __restrict__ W,
                                                    const float*  __restrict__ bias,
                                                    float* __restrict__ out) {
    constexpr int LST = K + 8;                    // LDS row stride in halfs (bank pad)
    constexpr int NK  = K >> 5;                   // WMMA K-steps
    constexpr int CPR = K >> 3;                   // 16B chunks per row
    extern __shared__ char smem[];
    bf16_t* sA = (bf16_t*)smem;                   // [128][LST]
    bf16_t* sW = sA + 128 * LST;                  // [256][LST]

    const int tid  = threadIdx.x;
    const int lane = tid & 31;
    const int wave = tid >> 5;
    const long long row0 = (long long)blockIdx.x * 128;

    // stage A tile (coalesced 16B chunks)
    for (int idx = tid; idx < 128 * CPR; idx += 256) {
        int r = idx / CPR, c = (idx % CPR) * 8;
        *(v8bf*)(sA + r * LST + c) = *(const v8bf*)(A + (row0 + r) * K + c);
    }
    // stage W tile
    for (int idx = tid; idx < 256 * CPR; idx += 256) {
        int r = idx / CPR, c = (idx % CPR) * 8;
        *(v8bf*)(sW + r * LST + c) = *(const v8bf*)(W + (long long)r * K + c);
    }
    __syncthreads();

    const int mrow = lane & 15;
    const int ksel = lane >> 4;                   // 0 or 1

    // A fragments for this wave's 16 rows, whole K, held in registers
    v16bf afrag[NK];
    #pragma unroll
    for (int kk = 0; kk < NK; ++kk) {
        const bf16_t* p = sA + (wave * 16 + mrow) * LST + kk * 32 + ksel * 8;
        v8bf lo = *(const v8bf*)(p);
        v8bf hi = *(const v8bf*)(p + 16);
        v16bf a;
        #pragma unroll
        for (int i = 0; i < 8; ++i) { a[i] = lo[i]; a[i + 8] = hi[i]; }
        afrag[kk] = a;
    }

    // sweep 16 N-tiles
    #pragma unroll 4
    for (int nt = 0; nt < 16; ++nt) {
        const int ncol = nt * 16 + mrow;
        const float bs = bias[ncol];
        v8f acc = { bs, bs, bs, bs, bs, bs, bs, bs };
        #pragma unroll
        for (int kk = 0; kk < NK; ++kk) {
            const bf16_t* p = sW + ncol * LST + kk * 32 + ksel * 8;
            v8bf lo = *(const v8bf*)(p);
            v8bf hi = *(const v8bf*)(p + 16);
            v16bf b;
            #pragma unroll
            for (int i = 0; i < 8; ++i) { b[i] = lo[i]; b[i + 8] = hi[i]; }
            acc = __builtin_amdgcn_wmma_f32_16x16x32_bf16(
                      false, afrag[kk], false, b, (short)0, acc, false, false);
        }
        #pragma unroll
        for (int r = 0; r < 8; ++r) {
            int m = wave * 16 + ksel * 8 + r;
            out[(row0 + m) * HID + ncol] = acc[r];
        }
    }
}

// ---------- recurrence: for t: h = tanh(xw[:,t,:] + h * Whh^T) ----------
// Block: 256 threads (8 waves), 32 batch rows. Whh + double-buffered h in LDS.
__global__ __launch_bounds__(256) void wmma_rnn_rec(const float*  __restrict__ xw,   // [B*T,256]
                                                    const bf16_t* __restrict__ Whh,  // [256,256]
                                                    bf16_t* __restrict__ hout) {     // [B*T,256]
    extern __shared__ char smem[];
    const int WST = HID + 8;                      // 264 halfs
    bf16_t* sW = (bf16_t*)smem;                   // [256][264]
    bf16_t* sH = sW + HID * WST;                  // [2][32][264]

    const int tid  = threadIdx.x;
    const int lane = tid & 31;
    const int wave = tid >> 5;
    const int mhalf = wave & 1;                   // which 16-row half of the 32-row tile
    const int ngrp  = wave >> 1;                  // which 64-col group
    const long long b0 = (long long)blockIdx.x * 32;

    // stage Whh
    for (int idx = tid; idx < (HID * HID) / 8; idx += 256) {
        int r = idx >> 5;                         // 32 chunks of 8 halfs per row
        int c = (idx & 31) * 8;
        *(v8bf*)(sW + r * WST + c) = *(const v8bf*)(Whh + r * HID + c);
    }
    // h(0) = 0
    for (int idx = tid; idx < 32 * WST; idx += 256) sH[idx] = (bf16_t)0.0f;
    __syncthreads();

    const int mrow = lane & 15;
    const int ksel = lane >> 4;

    int cur = 0;
    for (int t = 0; t < SEQT; ++t) {
        bf16_t* hcur = sH + cur * 32 * WST;
        bf16_t* hnxt = sH + (cur ^ 1) * 32 * WST;

        // acc <- xw tile
        v8f acc[4];
        #pragma unroll
        for (int nt = 0; nt < 4; ++nt) {
            const int ncol = ngrp * 64 + nt * 16 + mrow;
            #pragma unroll
            for (int r = 0; r < 8; ++r) {
                int m = mhalf * 16 + ksel * 8 + r;
                acc[nt][r] = xw[((b0 + m) * SEQT + t) * HID + ncol];
            }
        }
        // K = 256 in 8 WMMA steps
        #pragma unroll
        for (int kk = 0; kk < 8; ++kk) {
            const bf16_t* pa = hcur + (mhalf * 16 + mrow) * WST + kk * 32 + ksel * 8;
            v8bf alo = *(const v8bf*)pa;
            v8bf ahi = *(const v8bf*)(pa + 16);
            v16bf a;
            #pragma unroll
            for (int i = 0; i < 8; ++i) { a[i] = alo[i]; a[i + 8] = ahi[i]; }
            #pragma unroll
            for (int nt = 0; nt < 4; ++nt) {
                const int ncol = ngrp * 64 + nt * 16 + mrow;
                const bf16_t* pb = sW + ncol * WST + kk * 32 + ksel * 8;
                v8bf blo = *(const v8bf*)pb;
                v8bf bhi = *(const v8bf*)(pb + 16);
                v16bf b;
                #pragma unroll
                for (int i = 0; i < 8; ++i) { b[i] = blo[i]; b[i + 8] = bhi[i]; }
                acc[nt] = __builtin_amdgcn_wmma_f32_16x16x32_bf16(
                              false, a, false, b, (short)0, acc[nt], false, false);
            }
        }
        // tanh, write next-state LDS buffer + global sequence output
        #pragma unroll
        for (int nt = 0; nt < 4; ++nt) {
            const int ncol = ngrp * 64 + nt * 16 + mrow;
            #pragma unroll
            for (int r = 0; r < 8; ++r) {
                int m = mhalf * 16 + ksel * 8 + r;
                bf16_t hv = f32_to_bf16(tanhf(acc[nt][r]));
                hnxt[m * WST + ncol] = hv;
                hout[((b0 + m) * SEQT + t) * HID + ncol] = hv;
            }
        }
        __syncthreads();
        cur ^= 1;
    }
}

// ---------- FC: out[b,c] = h2[b,:] . fc_w[c,:] + fc_b[c] ----------
__global__ __launch_bounds__(256) void fc_kernel(const bf16_t* __restrict__ h2,   // [B,8192]
                                                 const bf16_t* __restrict__ fw,   // [12,8192]
                                                 const float*  __restrict__ fb,
                                                 float* __restrict__ out) {
    const int lane = threadIdx.x & 31;
    const int wave = threadIdx.x >> 5;
    const long long row = (long long)blockIdx.x * 8 + wave;
    if (row >= NBATCH) return;
    const bf16_t* hr = h2 + row * (HID * SEQT);

    float acc[NCLS];
    #pragma unroll
    for (int c = 0; c < NCLS; ++c) acc[c] = 0.0f;

    for (int k = lane; k < HID * SEQT; k += 32) {
        float hv = (float)hr[k];
        #pragma unroll
        for (int c = 0; c < NCLS; ++c)
            acc[c] += hv * (float)fw[c * (HID * SEQT) + k];
    }
    #pragma unroll
    for (int c = 0; c < NCLS; ++c)
        for (int off = 16; off > 0; off >>= 1)
            acc[c] += __shfl_xor(acc[c], off, 32);
    if (lane == 0) {
        #pragma unroll
        for (int c = 0; c < NCLS; ++c) out[row * NCLS + c] = acc[c] + fb[c];
    }
}

// ---------- host ----------
static inline size_t align_up(size_t x, size_t a) { return (x + a - 1) & ~(a - 1); }

extern "C" void kernel_launch(void* const* d_in, const int* in_sizes, int n_in,
                              void* d_out, int out_size, void* d_ws, size_t ws_size,
                              hipStream_t stream) {
    const float* x     = (const float*)d_in[0];   // [B,1,32,32]
    const float* w_ih0 = (const float*)d_in[1];   // [256,32]
    const float* w_hh0 = (const float*)d_in[2];   // [256,256]
    const float* b_ih0 = (const float*)d_in[3];
    const float* b_hh0 = (const float*)d_in[4];
    const float* w_ih1 = (const float*)d_in[5];   // [256,256]
    const float* w_hh1 = (const float*)d_in[6];   // [256,256]
    const float* b_ih1 = (const float*)d_in[7];
    const float* b_hh1 = (const float*)d_in[8];
    const float* fc_w  = (const float*)d_in[9];   // [12,8192]
    const float* fc_b  = (const float*)d_in[10];
    float* out = (float*)d_out;

    // workspace carve-out
    char* ws = (char*)d_ws;
    size_t off = 0;
    float*  xw     = (float*)(ws + off);  off = align_up(off + (size_t)NROWS * HID * 4, 256);
    bf16_t* h1     = (bf16_t*)(ws + off); off = align_up(off + (size_t)NROWS * HID * 2, 256);
    bf16_t* h2     = (bf16_t*)(ws + off); off = align_up(off + (size_t)NROWS * HID * 2, 256);
    bf16_t* x_bf   = (bf16_t*)(ws + off); off = align_up(off + (size_t)NROWS * 32 * 2, 256);
    bf16_t* wih0_b = (bf16_t*)(ws + off); off = align_up(off + 256 * 32 * 2, 256);
    bf16_t* whh0_b = (bf16_t*)(ws + off); off = align_up(off + 256 * 256 * 2, 256);
    bf16_t* wih1_b = (bf16_t*)(ws + off); off = align_up(off + 256 * 256 * 2, 256);
    bf16_t* whh1_b = (bf16_t*)(ws + off); off = align_up(off + 256 * 256 * 2, 256);
    bf16_t* fcw_b  = (bf16_t*)(ws + off); off = align_up(off + (size_t)NCLS * HID * SEQT * 2, 256);
    float*  bias0  = (float*)(ws + off);  off = align_up(off + 256 * 4, 256);
    float*  bias1  = (float*)(ws + off);  off = align_up(off + 256 * 4, 256);
    (void)ws_size; (void)n_in; (void)in_sizes; (void)out_size;

    // 1) precision conversion + bias folding
    {
        int n = NROWS * 32;  // x
        cvt_f32_bf16<<<(n + 255) / 256, 256, 0, stream>>>(x, x_bf, n);
        n = 256 * 32;
        cvt_f32_bf16<<<(n + 255) / 256, 256, 0, stream>>>(w_ih0, wih0_b, n);
        n = 256 * 256;
        cvt_f32_bf16<<<(n + 255) / 256, 256, 0, stream>>>(w_hh0, whh0_b, n);
        cvt_f32_bf16<<<(n + 255) / 256, 256, 0, stream>>>(w_ih1, wih1_b, n);
        cvt_f32_bf16<<<(n + 255) / 256, 256, 0, stream>>>(w_hh1, whh1_b, n);
        n = NCLS * HID * SEQT;
        cvt_f32_bf16<<<(n + 255) / 256, 256, 0, stream>>>(fc_w, fcw_b, n);
        add_bias_vec<<<1, 256, 0, stream>>>(b_ih0, b_hh0, bias0, 256);
        add_bias_vec<<<1, 256, 0, stream>>>(b_ih1, b_hh1, bias1, 256);
    }

    const int gemm_blocks = NROWS / 128;           // 4096
    const int rec_blocks  = NBATCH / 32;           // 512
    const size_t smem_g32  = (size_t)(128 + 256) * (32 + 8) * 2;    //  30.7 KB
    const size_t smem_g256 = (size_t)(128 + 256) * (256 + 8) * 2;   // 202.8 KB
    const size_t smem_rec  = (size_t)(256 + 64) * (256 + 8) * 2;    // 169.0 KB

    // 2) layer 0: xw = x * w_ih0^T + bias0 ; recurrence -> h1
    wmma_gemm_xw<32><<<gemm_blocks, 256, smem_g32, stream>>>(x_bf, wih0_b, bias0, xw);
    wmma_rnn_rec<<<rec_blocks, 256, smem_rec, stream>>>(xw, whh0_b, h1);

    // 3) layer 1: xw = h1 * w_ih1^T + bias1 ; recurrence -> h2
    wmma_gemm_xw<256><<<gemm_blocks, 256, smem_g256, stream>>>(h1, wih1_b, bias1, xw);
    wmma_rnn_rec<<<rec_blocks, 256, smem_rec, stream>>>(xw, whh1_b, h2);

    // 4) classifier
    fc_kernel<<<NBATCH / 8, 256, 0, stream>>>(h2, fcw_b, fc_b, out);
}
